// VQVAE_6150393168134
// MI455X (gfx1250) — compile-verified, matrix operand order checked by reading, not run on hardware
//
#include <hip/hip_runtime.h>
#include <hip/hip_bf16.h>
#include <stdint.h>

typedef unsigned short u16;
typedef __bf16  bf16x16 __attribute__((ext_vector_type(16)));
typedef float   v8f     __attribute__((ext_vector_type(8)));
typedef unsigned int u32x4 __attribute__((ext_vector_type(4)));
typedef int v4i_gnu __attribute__((vector_size(16)));

#define BSZ 1024
#define TLEN 128
#define ENH 256
#define ZD 32
#define KCODE 512
#define VOC 51

// d_out layout: [0]=xloss, [1]=nonpadded, [2..2+B*127)=pred, [130050]=Q, [130051..]=idx
#define PRED_BASE 2
#define Q_OFF (2 + BSZ * 127)
#define IDX_OFF (Q_OFF + 1)

__device__ __forceinline__ u16 f2bf(float f) {
    union { float f; uint32_t u; } x; x.f = f;
    uint32_t u = x.u;
    uint32_t r = ((u >> 16) & 1u) + 0x7FFFu;
    return (u16)((u + r) >> 16);
}

union FragU { bf16x16 v; u32x4 q[2]; };

// CDNA5 16-bit fragment: elems 0..7 = cols kb..kb+7, elems 8..15 = cols kb+16..kb+23
__device__ __forceinline__ bf16x16 load_frag(const u16* p) {
    FragU u;
    u.q[0] = *(const u32x4*)(p);
    u.q[1] = *(const u32x4*)(p + 16);
    return u.v;
}

// C[M,N] = [A1 | A2] (bf16, row-major) x W[N, K1+K2]^T (bf16, row-major) + bias
// Block = 256 threads = 8 waves: 128 rows x 64 cols. W panel staged in LDS via
// gfx1250 async global->LDS copies. M mult of 128, N mult of 64, K1/K2 mult of 32.
__global__ void gemm_wmma(const u16* __restrict__ A1, int lds1, int K1,
                          const u16* __restrict__ A2, int lds2, int K2,
                          const u16* __restrict__ W, const float* __restrict__ bias,
                          float* __restrict__ C, int ldc, int M, int N) {
    extern __shared__ u16 sW[];                  // [64][Kcat]
    const int Kcat   = K1 + K2;
    const int tilesN = N >> 6;
    const int bm     = blockIdx.x / tilesN;      // block row tile (128 rows)
    const int tn     = (blockIdx.x % tilesN) << 6;
    const int tid    = threadIdx.x;

    // ---- stage W panel rows [tn, tn+64) into LDS ----
    {
        const char* gsrc = (const char*)(W + (size_t)tn * Kcat);
        char* ldst = (char*)sW;
        const int panelBytes = 64 * Kcat * 2;
#if defined(__gfx1250__) && __has_builtin(__builtin_amdgcn_global_load_async_to_lds_b128)
        typedef __attribute__((address_space(1))) v4i_gnu gv4i;
        typedef __attribute__((address_space(3))) v4i_gnu lv4i;
        for (int off = tid * 16; off < panelBytes; off += 256 * 16) {
            __builtin_amdgcn_global_load_async_to_lds_b128(
                (gv4i*)(gsrc + off), (lv4i*)(ldst + off), 0, 0);
        }
#if __has_builtin(__builtin_amdgcn_s_wait_asynccnt)
        __builtin_amdgcn_s_wait_asynccnt(0);
#else
        asm volatile("s_wait_asynccnt 0x0" ::: "memory");
#endif
#else
        for (int off = tid * 16; off < panelBytes; off += 256 * 16)
            *(u32x4*)(ldst + off) = *(const u32x4*)(gsrc + off);
#endif
    }
    __syncthreads();

    const int wave = tid >> 5;
    const int lane = tid & 31;
    const int tm   = (bm << 7) + (wave << 4);
    const int r    = lane & 15;
    const int hi   = (lane >> 4) ? 8 : 0;        // K sub-offset for A/B, row sub-offset for C

    v8f acc0 = {}, acc1 = {}, acc2 = {}, acc3 = {};
    const u16* a1row = A1 + (size_t)(tm + r) * lds1;
    const u16* a2row = A2 ? (A2 + (size_t)(tm + r) * lds2) : nullptr;

    for (int k0 = 0; k0 < K1; k0 += 32) {
        int kb = k0 + hi;
        bf16x16 a  = load_frag(a1row + kb);
        bf16x16 b0 = load_frag(sW + (size_t)(r +  0) * Kcat + kb);
        bf16x16 b1 = load_frag(sW + (size_t)(r + 16) * Kcat + kb);
        bf16x16 b2 = load_frag(sW + (size_t)(r + 32) * Kcat + kb);
        bf16x16 b3 = load_frag(sW + (size_t)(r + 48) * Kcat + kb);
        acc0 = __builtin_amdgcn_wmma_f32_16x16x32_bf16(false, a, false, b0, (short)0, acc0, false, false);
        acc1 = __builtin_amdgcn_wmma_f32_16x16x32_bf16(false, a, false, b1, (short)0, acc1, false, false);
        acc2 = __builtin_amdgcn_wmma_f32_16x16x32_bf16(false, a, false, b2, (short)0, acc2, false, false);
        acc3 = __builtin_amdgcn_wmma_f32_16x16x32_bf16(false, a, false, b3, (short)0, acc3, false, false);
    }
    for (int k0 = 0; k0 < K2; k0 += 32) {
        int kb = k0 + hi;           // segment-local col for A2
        int kk = K1 + kb;           // global col for W
        bf16x16 a  = load_frag(a2row + kb);
        bf16x16 b0 = load_frag(sW + (size_t)(r +  0) * Kcat + kk);
        bf16x16 b1 = load_frag(sW + (size_t)(r + 16) * Kcat + kk);
        bf16x16 b2 = load_frag(sW + (size_t)(r + 32) * Kcat + kk);
        bf16x16 b3 = load_frag(sW + (size_t)(r + 48) * Kcat + kk);
        acc0 = __builtin_amdgcn_wmma_f32_16x16x32_bf16(false, a, false, b0, (short)0, acc0, false, false);
        acc1 = __builtin_amdgcn_wmma_f32_16x16x32_bf16(false, a, false, b1, (short)0, acc1, false, false);
        acc2 = __builtin_amdgcn_wmma_f32_16x16x32_bf16(false, a, false, b2, (short)0, acc2, false, false);
        acc3 = __builtin_amdgcn_wmma_f32_16x16x32_bf16(false, a, false, b3, (short)0, acc3, false, false);
    }

    int c0 = tn + r;
    float bb0 = bias ? bias[c0 +  0] : 0.f;
    float bb1 = bias ? bias[c0 + 16] : 0.f;
    float bb2 = bias ? bias[c0 + 32] : 0.f;
    float bb3 = bias ? bias[c0 + 48] : 0.f;
    #pragma unroll
    for (int v = 0; v < 8; ++v) {
        float* crow = C + (size_t)(tm + v + hi) * ldc;
        crow[c0 +  0] = acc0[v] + bb0;
        crow[c0 + 16] = acc1[v] + bb1;
        crow[c0 + 32] = acc2[v] + bb2;
        crow[c0 + 48] = acc3[v] + bb3;
    }
}

// Build fused GRU weight [1024, Kh+Kx] and bias [1024]:
// rows 0..511: (Whh|Wih) rows 0..511 (r,z gates fused); rows 512..767: hn (h-part);
// rows 768..1023: inn (x-part).
__global__ void build_gru_w(u16* __restrict__ dW, float* __restrict__ dB,
                            const float* __restrict__ Wih, const float* __restrict__ Whh,
                            const float* __restrict__ bih, const float* __restrict__ bhh,
                            int Kx, int Kh) {
    int Kcat = Kh + Kx;
    int idx = blockIdx.x * 256 + threadIdx.x;
    int total = 1024 * Kcat;
    if (idx < total) {
        int n = idx / Kcat, k = idx % Kcat;
        float v = 0.f;
        if (n < 512) {
            v = (k < Kh) ? Whh[(size_t)n * Kh + k] : Wih[(size_t)n * Kx + (k - Kh)];
        } else if (n < 768) {
            if (k < Kh) v = Whh[(size_t)n * Kh + k];
        } else {
            if (k >= Kh) v = Wih[(size_t)(n - 256) * Kx + (k - Kh)];
        }
        dW[idx] = f2bf(v);
    }
    if (idx < 1024) {
        float bv;
        if (idx < 512)      bv = bih[idx] + bhh[idx];
        else if (idx < 768) bv = bhh[idx];
        else                bv = bih[idx - 256];
        dB[idx] = bv;
    }
}

__global__ void conv_pad(u16* __restrict__ dst, const float* __restrict__ src,
                         int N, int K, int dN, int dK) {
    int idx = blockIdx.x * 256 + threadIdx.x;
    if (idx >= dN * dK) return;
    int n = idx / dK, k = idx % dK;
    dst[idx] = (n < N && k < K) ? f2bf(src[(size_t)n * K + k]) : (u16)0;
}

__global__ void pad_bias(float* __restrict__ dst, const float* __restrict__ src, int N, int dN) {
    int idx = blockIdx.x * 256 + threadIdx.x;
    if (idx < dN) dst[idx] = (idx < N) ? src[idx] : 0.f;
}

__global__ void zero_f32_k(float* p, int n) {
    int i = blockIdx.x * 256 + threadIdx.x;
    if (i < n) p[i] = 0.f;
}

__global__ void init_out_k(float* out) {
    if (threadIdx.x == 0) { out[0] = 0.f; out[1] = 0.f; out[Q_OFF] = 0.f; }
}

__global__ void embed_enc_k(u16* __restrict__ e, const float* __restrict__ emb,
                            const int* __restrict__ x) {
    int i = blockIdx.x * 256 + threadIdx.x;           // i = (t*B + b)*64 + k
    if (i >= TLEN * BSZ * 64) return;
    int k = i & 63; int rb = i >> 6; int b = rb & (BSZ - 1); int t = rb >> 10;
    e[i] = f2bf(emb[(size_t)x[b * TLEN + t] * 64 + k]);
}

__global__ void build_dec_in_k(u16* __restrict__ din, const u16* __restrict__ qzbf,
                               const float* __restrict__ emb, const int* __restrict__ x) {
    int i = blockIdx.x * 256 + threadIdx.x;           // i = (t*B + b)*96 + k
    if (i >= 127 * BSZ * 96) return;
    int k = i % 96; int rb = i / 96; int b = rb & (BSZ - 1); int t = rb >> 10;
    u16 v;
    if (k < 32) v = qzbf[b * 32 + k];
    else        v = f2bf(emb[(size_t)x[b * TLEN + t] * 64 + (k - 32)]);
    din[i] = v;
}

// gates g[B,1024]: [r | z | hn | inn]; h update in f32 + bf16 copy (+ optional ys collect)
__global__ void gru_update(const float* __restrict__ g, float* __restrict__ h,
                           u16* __restrict__ hbf, u16* __restrict__ ysbf) {
    int i = blockIdx.x * 256 + threadIdx.x;
    if (i >= BSZ * ENH) return;
    int b = i >> 8, j = i & 255;
    const float* gb = g + (size_t)b * 1024;
    float r  = 1.f / (1.f + __expf(-gb[j]));
    float zg = 1.f / (1.f + __expf(-gb[256 + j]));
    float n  = tanhf(gb[768 + j] + r * gb[512 + j]);
    float hn = (1.f - zg) * n + zg * h[i];
    h[i] = hn;
    u16 hb = f2bf(hn);
    hbf[i] = hb;
    if (ysbf) ysbf[i] = hb;
}

__global__ void f32_to_bf16_k(u16* dst, const float* src, int n) {
    int i = blockIdx.x * 256 + threadIdx.x;
    if (i < n) dst[i] = f2bf(src[i]);
}

__global__ void vq_k(const float* __restrict__ zbuf, const float* __restrict__ codebook,
                     u16* __restrict__ qzbf, float* __restrict__ out) {
    __shared__ float zs[ZD];
    __shared__ float best[256];
    __shared__ int   bidx[256];
    int b = blockIdx.x, t = threadIdx.x;
    if (t < ZD) zs[t] = zbuf[b * 64 + t];
    __syncthreads();
    float bd = 3.4e38f; int bi = 0;
    for (int k = t; k < KCODE; k += 256) {
        const float* c = codebook + (size_t)k * ZD;
        float d = 0.f;
        #pragma unroll
        for (int j = 0; j < ZD; ++j) { float df = zs[j] - c[j]; d += df * df; }
        if (d < bd) { bd = d; bi = k; }
    }
    best[t] = bd; bidx[t] = bi;
    __syncthreads();
    for (int s = 128; s > 0; s >>= 1) {
        if (t < s) {
            if (best[t + s] < best[t] || (best[t + s] == best[t] && bidx[t + s] < bidx[t])) {
                best[t] = best[t + s]; bidx[t] = bidx[t + s];
            }
        }
        __syncthreads();
    }
    int w = bidx[0];
    if (t < ZD) qzbf[b * ZD + t] = f2bf(codebook[(size_t)w * ZD + t]);
    if (t == 0) {
        out[IDX_OFF + b] = (float)w;
        float s = 0.f;
        #pragma unroll
        for (int j = 0; j < ZD; ++j) { float df = codebook[(size_t)w * ZD + j] - zs[j]; s += df * df; }
        atomicAdd(&out[Q_OFF], 0.2f * s / (float)ZD);   // 0.1*commit + 0.1*embed
    }
}

__global__ void loss_k(const float* __restrict__ logits, const int* __restrict__ x,
                       float* __restrict__ out) {
    int i = blockIdx.x * 256 + threadIdx.x;           // i = t*B + b
    if (i >= 127 * BSZ) return;
    int b = i & (BSZ - 1), t = i >> 10;
    const float* lr = logits + (size_t)i * 64;
    float mx = lr[0]; int am = 0;
    for (int v = 1; v < VOC; ++v) { float lv = lr[v]; if (lv > mx) { mx = lv; am = v; } }
    float se = 0.f;
    for (int v = 0; v < VOC; ++v) se += __expf(lr[v] - mx);
    float lse = mx + __logf(se);
    int tgt = x[b * TLEN + (t + 1)];
    out[PRED_BASE + b * 127 + t] = (float)am;
    if (tgt != 0) {
        atomicAdd(&out[0], lse - lr[tgt]);
        atomicAdd(&out[1], 1.0f);
    }
}

static inline void launch_gemm(const u16* A1, int lds1, int K1, const u16* A2, int lds2, int K2,
                               const u16* W, const float* bias, float* C, int ldc,
                               int M, int N, hipStream_t s) {
    int tilesN = N / 64;
    int blocks = (M / 128) * tilesN;
    size_t shm = (size_t)64 * (K1 + K2) * sizeof(u16);
    gemm_wmma<<<blocks, 256, shm, s>>>(A1, lds1, K1, A2, lds2, K2, W, bias, C, ldc, M, N);
}

extern "C" void kernel_launch(void* const* d_in, const int* in_sizes, int n_in,
                              void* d_out, int out_size, void* d_ws, size_t ws_size,
                              hipStream_t stream) {
    (void)in_sizes; (void)n_in; (void)out_size; (void)ws_size;
    const int*   x        = (const int*)d_in[0];
    const float* emb_enc  = (const float*)d_in[1];
    const float* emb_dec  = (const float*)d_in[2];
    const float* Wih_f = (const float*)d_in[3],  *Whh_f = (const float*)d_in[4];
    const float* bih_f = (const float*)d_in[5],  *bhh_f = (const float*)d_in[6];
    const float* Wih_b = (const float*)d_in[7],  *Whh_b = (const float*)d_in[8];
    const float* bih_b = (const float*)d_in[9],  *bhh_b = (const float*)d_in[10];
    const float* Wdown = (const float*)d_in[11], *bdown = (const float*)d_in[12];
    const float* codebook = (const float*)d_in[13];
    const float* Wup   = (const float*)d_in[14], *bup   = (const float*)d_in[15];
    const float* Wih_d = (const float*)d_in[16], *Whh_d = (const float*)d_in[17];
    const float* bih_d = (const float*)d_in[18], *bhh_d = (const float*)d_in[19];
    const float* Wout  = (const float*)d_in[20], *bout  = (const float*)d_in[21];
    float* out = (float*)d_out;

    // ---- workspace bump allocator ----
    char* wp = (char*)d_ws;
    auto alloc = [&](size_t bytes) -> char* {
        char* p = wp; wp += (bytes + 255) & ~(size_t)255; return p;
    };
    u16*   Wcat_f  = (u16*)alloc(1024 * 320 * sizeof(u16));
    u16*   Wcat_b  = (u16*)alloc(1024 * 320 * sizeof(u16));
    u16*   Wcat_d  = (u16*)alloc(1024 * 352 * sizeof(u16));
    float* bias_f  = (float*)alloc(1024 * 4);
    float* bias_b  = (float*)alloc(1024 * 4);
    float* bias_d  = (float*)alloc(1024 * 4);
    u16*   Wdn_bf  = (u16*)alloc(64 * 512 * sizeof(u16));
    float* bdn_p   = (float*)alloc(64 * 4);
    u16*   Wup_bf  = (u16*)alloc(256 * 32 * sizeof(u16));
    u16*   Wout_bf = (u16*)alloc(64 * 256 * sizeof(u16));
    float* bout_p  = (float*)alloc(64 * 4);
    u16*   e_enc   = (u16*)alloc((size_t)TLEN * BSZ * 64 * sizeof(u16));
    u16*   dec_in  = (u16*)alloc((size_t)127 * BSZ * 96 * sizeof(u16));
    float* hf      = (float*)alloc((size_t)BSZ * ENH * 4);
    u16*   hf_bf   = (u16*)alloc((size_t)BSZ * ENH * sizeof(u16));
    float* hb      = (float*)alloc((size_t)BSZ * ENH * 4);
    u16*   hb_bf   = (u16*)alloc((size_t)BSZ * ENH * sizeof(u16));
    float* hd      = (float*)alloc((size_t)BSZ * ENH * 4);
    u16*   hd_bf   = (u16*)alloc((size_t)BSZ * ENH * sizeof(u16));
    float* gbuf    = (float*)alloc((size_t)BSZ * 1024 * 4);
    float* zbuf    = (float*)alloc((size_t)BSZ * 64 * 4);
    u16*   qz_bf   = (u16*)alloc((size_t)BSZ * ZD * sizeof(u16));
    u16*   ys_bf   = (u16*)alloc((size_t)127 * BSZ * ENH * sizeof(u16));
    float* logits  = (float*)alloc((size_t)127 * BSZ * 64 * 4);

    // ---- init ----
    init_out_k<<<1, 64, 0, stream>>>(out);
    zero_f32_k<<<(BSZ * ENH + 255) / 256, 256, 0, stream>>>(hf, BSZ * ENH);
    zero_f32_k<<<(BSZ * ENH + 255) / 256, 256, 0, stream>>>(hb, BSZ * ENH);
    zero_f32_k<<<(BSZ * ENH / 2 + 255) / 256, 256, 0, stream>>>((float*)hf_bf, BSZ * ENH / 2);
    zero_f32_k<<<(BSZ * ENH / 2 + 255) / 256, 256, 0, stream>>>((float*)hb_bf, BSZ * ENH / 2);

    // ---- weight prep (bf16, fused / padded) ----
    build_gru_w<<<(1024 * 320 + 255) / 256, 256, 0, stream>>>(Wcat_f, bias_f, Wih_f, Whh_f, bih_f, bhh_f, 64, 256);
    build_gru_w<<<(1024 * 320 + 255) / 256, 256, 0, stream>>>(Wcat_b, bias_b, Wih_b, Whh_b, bih_b, bhh_b, 64, 256);
    build_gru_w<<<(1024 * 352 + 255) / 256, 256, 0, stream>>>(Wcat_d, bias_d, Wih_d, Whh_d, bih_d, bhh_d, 96, 256);
    conv_pad<<<(64 * 512 + 255) / 256, 256, 0, stream>>>(Wdn_bf, Wdown, 32, 512, 64, 512);
    pad_bias<<<1, 256, 0, stream>>>(bdn_p, bdown, 32, 64);
    conv_pad<<<(256 * 32 + 255) / 256, 256, 0, stream>>>(Wup_bf, Wup, 256, 32, 256, 32);
    conv_pad<<<(64 * 256 + 255) / 256, 256, 0, stream>>>(Wout_bf, Wout, 51, 256, 64, 256);
    pad_bias<<<1, 256, 0, stream>>>(bout_p, bout, 51, 64);

    // ---- encoder embeddings ----
    embed_enc_k<<<(TLEN * BSZ * 64 + 255) / 256, 256, 0, stream>>>(e_enc, emb_enc, x);

    // ---- bidirectional encoder GRUs ----
    for (int t = 0; t < TLEN; ++t) {
        launch_gemm(hf_bf, 256, 256, e_enc + (size_t)t * BSZ * 64, 64, 64,
                    Wcat_f, bias_f, gbuf, 1024, BSZ, 1024, stream);
        gru_update<<<(BSZ * ENH + 255) / 256, 256, 0, stream>>>(gbuf, hf, hf_bf, nullptr);
    }
    for (int t = 0; t < TLEN; ++t) {
        launch_gemm(hb_bf, 256, 256, e_enc + (size_t)(TLEN - 1 - t) * BSZ * 64, 64, 64,
                    Wcat_b, bias_b, gbuf, 1024, BSZ, 1024, stream);
        gru_update<<<(BSZ * ENH + 255) / 256, 256, 0, stream>>>(gbuf, hb, hb_bf, nullptr);
    }

    // ---- z = [hf|hb] @ Wdown.T + bdown  (two-segment A) ----
    launch_gemm(hf_bf, 256, 256, hb_bf, 256, 256, Wdn_bf, bdn_p, zbuf, 64, BSZ, 64, stream);

    // ---- VQ: idx, Q loss, quantized (straight-through) ----
    vq_k<<<BSZ, 256, 0, stream>>>(zbuf, codebook, qz_bf, out);

    // ---- hidden = qz @ Wup.T + bup ----
    launch_gemm(qz_bf, 32, 32, nullptr, 0, 0, Wup_bf, bup, hd, 256, BSZ, 256, stream);
    f32_to_bf16_k<<<(BSZ * ENH + 255) / 256, 256, 0, stream>>>(hd_bf, hd, BSZ * ENH);

    // ---- decoder inputs [t][b][ qz(32) | emb_dec(64) ] ----
    build_dec_in_k<<<(127 * BSZ * 96 + 255) / 256, 256, 0, stream>>>(dec_in, qz_bf, emb_dec, x);

    // ---- decoder GRU (collect ys) ----
    for (int t = 0; t < 127; ++t) {
        launch_gemm(hd_bf, 256, 256, dec_in + (size_t)t * BSZ * 96, 96, 96,
                    Wcat_d, bias_d, gbuf, 1024, BSZ, 1024, stream);
        gru_update<<<(BSZ * ENH + 255) / 256, 256, 0, stream>>>(gbuf, hd, hd_bf,
                                                                ys_bf + (size_t)t * BSZ * ENH);
    }

    // ---- logits = ys @ Wout.T + bout (N padded to 64) ----
    launch_gemm(ys_bf, 256, 256, nullptr, 0, 0, Wout_bf, bout_p, logits, 64, 127 * BSZ, 64, stream);

    // ---- log-softmax NLL, argmax, pack outputs ----
    loss_k<<<(127 * BSZ + 255) / 256, 256, 0, stream>>>(logits, x, out);
}
